// CrossAttention_18459769439007
// MI455X (gfx1250) — compile-verified, matrix operand order checked by reading, not run on hardware
//
#include <hip/hip_runtime.h>
#include <cmath>

typedef __bf16 bf16_t;
typedef __attribute__((ext_vector_type(16))) __bf16 v16bf;
typedef __attribute__((ext_vector_type(8)))  __bf16 v8bf;
typedef __attribute__((ext_vector_type(8)))  float  v8f;
typedef __attribute__((ext_vector_type(4)))  int    v4i;

#define FLAG_OUT_BF16 1
#define FLAG_TRANS    2
#define FLAG_ACCUM    4

#if defined(__has_builtin)
#if __has_builtin(__builtin_amdgcn_global_load_async_to_lds_b128) && \
    __has_builtin(__builtin_amdgcn_s_wait_asynccnt)
#define HAVE_ASYNC_LDS 1
#endif
#endif

#ifdef HAVE_ASYNC_LDS
typedef __attribute__((address_space(1))) v4i as1_v4i;
typedef __attribute__((address_space(3))) v4i as3_v4i;
#endif

// Fragment helper: CDNA5 16-bit 16x32 A/B fragment. For lane L (hf = L>>4),
// element e holds K = (e&7) + 8*hf + 16*(e>>3). With p pre-offset by (kk + 8*hf),
// that is two contiguous 8-element (16-byte) chunks at p and p+16.
__device__ __forceinline__ v16bf load_frag_bf16(const bf16_t* p) {
  v8bf lo = *(const v8bf*)(p);
  v8bf hi = *(const v8bf*)(p + 16);
  v16bf r;
#pragma unroll
  for (int i = 0; i < 8; ++i) { r[i] = lo[i]; r[i + 8] = hi[i]; }
  return r;
}

__device__ __forceinline__ v16bf frag_from_f32(const float* p) {
  v8f lo = *(const v8f*)(p);
  v8f hi = *(const v8f*)(p + 16);
  v16bf r;
#pragma unroll
  for (int i = 0; i < 8; ++i) { r[i] = (bf16_t)lo[i]; r[i + 8] = (bf16_t)hi[i]; }
  return r;
}

__device__ __forceinline__ v8f wmma_bf16(v16bf a, v16bf b, v8f c) {
  return __builtin_amdgcn_wmma_f32_16x16x32_bf16(false, a, false, b, (short)0, c,
                                                 false, false);
}

// ---------------------------------------------------------------------------
// Pack kernels (f32 -> bf16)
// ---------------------------------------------------------------------------
__global__ __launch_bounds__(256) void nchw_to_nhwc_bf16(
    const float* __restrict__ src, bf16_t* __restrict__ dst, int C, int HW) {
  const long b = blockIdx.y;
  const long i = (long)blockIdx.x * 256 + threadIdx.x;  // over C*HW
  const int c = (int)(i % C);
  const long hw = i / C;
  dst[(b * HW + hw) * C + c] = (bf16_t)src[(b * C + c) * HW + hw];
}

// dst[r*C + c] = src[c*R + r]   (transpose-convert)
__global__ __launch_bounds__(256) void w_transpose_bf16(
    const float* __restrict__ src, bf16_t* __restrict__ dst, int R, int C) {
  const long i = (long)blockIdx.x * 256 + threadIdx.x;
  const int c = (int)(i % C);
  const int r = (int)(i / C);
  dst[(long)r * C + c] = (bf16_t)src[(long)c * R + r];
}

// dst[r*C + c] = src[r*ld + off + c]  (slice-convert)
__global__ __launch_bounds__(256) void w_convert_bf16(
    const float* __restrict__ src, bf16_t* __restrict__ dst,
    int R, int C, int ld, int off) {
  const long i = (long)blockIdx.x * 256 + threadIdx.x;
  const int c = (int)(i % C);
  const int r = (int)(i / C);
  dst[(long)r * C + c] = (bf16_t)src[(long)r * ld + off + c];
}

// ---------------------------------------------------------------------------
// Generic batched WMMA GEMM: C[m,n] = sum_k A[m,k] * Bt[n,k] + bias[n]
// A: [M,K] bf16 row-major, Bt: [N,K] bf16 row-major.
// Block = 256 threads = 8 wave32 arranged 2(M) x 4(N).
// Wave tile = 64x64 (4x4 accumulators); block tile = 128(M) x 256(N).
// 16 WMMAs per 8 fragment loads per k-step (4x register reuse of A and B).
// ---------------------------------------------------------------------------
__global__ __launch_bounds__(256) void gemm_bf16_kernel(
    const bf16_t* __restrict__ A, const bf16_t* __restrict__ Bt,
    const float* __restrict__ bias, void* __restrict__ Cout,
    int M, int N, int K, long sA, long sB, long sC, int ldc, int flags) {
  const int wv = threadIdx.x >> 5;
  const int lane = threadIdx.x & 31;
  const int hf = lane >> 4;
  const int lr = lane & 15;
  const int wm = wv >> 2;          // 0..1
  const int wn = wv & 3;           // 0..3
  const int m_base = (blockIdx.y << 7) + wm * 64;
  const int n_base = (blockIdx.x << 8) + wn * 64;
  const long b = blockIdx.z;

  const bf16_t* pa = A + b * sA + (long)(m_base + lr) * K + 8 * hf;
  const bf16_t* pb = Bt + b * sB + (long)(n_base + lr) * K + 8 * hf;

  v8f acc[4][4];
#pragma unroll
  for (int i = 0; i < 4; ++i)
#pragma unroll
    for (int j = 0; j < 4; ++j) acc[i][j] = (v8f){};

#pragma unroll 2
  for (int kk = 0; kk < K; kk += 32) {
    __builtin_prefetch(pa + kk + 64, 0, 0);   // -> global_prefetch_b8
    __builtin_prefetch(pb + kk + 64, 0, 0);
    v16bf af[4], bf[4];
#pragma unroll
    for (int i = 0; i < 4; ++i) af[i] = load_frag_bf16(pa + (long)i * 16 * K + kk);
#pragma unroll
    for (int j = 0; j < 4; ++j) bf[j] = load_frag_bf16(pb + (long)j * 16 * K + kk);
#pragma unroll
    for (int i = 0; i < 4; ++i)
#pragma unroll
      for (int j = 0; j < 4; ++j) acc[i][j] = wmma_bf16(af[i], bf[j], acc[i][j]);
  }

  float bval[4];
#pragma unroll
  for (int j = 0; j < 4; ++j)
    bval[j] = bias ? bias[n_base + j * 16 + lr] : 0.0f;

#pragma unroll
  for (int i = 0; i < 4; ++i) {
#pragma unroll
    for (int j = 0; j < 4; ++j) {
      const int col = n_base + j * 16 + lr;
#pragma unroll
      for (int r = 0; r < 8; ++r) {
        const int row = m_base + i * 16 + r + 8 * hf;  // C/D: vgpr r, hf -> M+8
        float v = acc[i][j][r] + bval[j];
        const long idx =
            b * sC + ((flags & FLAG_TRANS) ? ((long)col * ldc + row)
                                           : ((long)row * ldc + col));
        if (flags & FLAG_OUT_BF16) {
          ((bf16_t*)Cout)[idx] = (bf16_t)v;
        } else {
          float* Cf = (float*)Cout;
          if (flags & FLAG_ACCUM) Cf[idx] += v;
          else Cf[idx] = v;
        }
      }
    }
  }
}

// ---------------------------------------------------------------------------
// Attention: one block (256 thr, 8 waves) per 16 query rows of one batch.
// Q: [b][4096][512] bf16, Km: [b][1024][512] bf16, Vt: [b][512][1024] bf16.
// O: [b][4096][512] bf16.
// Dynamic LDS: Qs 16x512 bf16 (16K) | S 16x1024 f32 (64K) | red 16x16 | stat 16.
// ---------------------------------------------------------------------------
__global__ __launch_bounds__(256) void attn_kernel(
    const bf16_t* __restrict__ Q, const bf16_t* __restrict__ Km,
    const bf16_t* __restrict__ Vt, bf16_t* __restrict__ O, float scale) {
  extern __shared__ char smem_raw[];
  bf16_t* Qs = (bf16_t*)smem_raw;                          // 16*512*2 = 16384
  float* S = (float*)(smem_raw + 16384);                   // 16*1024*4 = 65536
  float* red = (float*)(smem_raw + 16384 + 65536);         // 16*16*4  = 1024
  float* stat = (float*)(smem_raw + 16384 + 65536 + 1024); // 16*4

  const long b = blockIdx.y;
  const int m0 = blockIdx.x << 4;
  const int tid = threadIdx.x;
  const int wv = tid >> 5;
  const int lane = tid & 31;
  const int hf = lane >> 4;
  const int lr = lane & 15;

  // 1. stage Q tile (16x512 bf16) into LDS
  {
    const uint4* src = (const uint4*)(Q + ((b * 4096) + m0) * 512);
    uint4* dst = (uint4*)Qs;
#ifdef HAVE_ASYNC_LDS
#pragma unroll
    for (int i = tid; i < 1024; i += 256)
      __builtin_amdgcn_global_load_async_to_lds_b128(
          (as1_v4i*)(src + i), (as3_v4i*)(dst + i), 0, 0);
    __builtin_amdgcn_s_wait_asynccnt(0);
#else
#pragma unroll
    for (int i = tid; i < 1024; i += 256) dst[i] = src[i];
#endif
  }
  __syncthreads();

  // 2. S = scale * Q K^T : k-outer, 8 accumulators per wave (A-frag reused 8x)
  {
    v8f accs[8];
#pragma unroll
    for (int t = 0; t < 8; ++t) accs[t] = (v8f){};
    const bf16_t* pq = Qs + (long)lr * 512 + 8 * hf;
    const bf16_t* pk0 = Km + (b * 1024 + wv * 16 + lr) * 512 + 8 * hf;
    for (int kk = 0; kk < 512; kk += 32) {
      v16bf af = load_frag_bf16(pq + kk);
#pragma unroll
      for (int t = 0; t < 8; ++t) {
        v16bf bf = load_frag_bf16(pk0 + (long)t * 128 * 512 + kk);
        accs[t] = wmma_bf16(af, bf, accs[t]);
      }
    }
#pragma unroll
    for (int t = 0; t < 8; ++t)
#pragma unroll
      for (int r = 0; r < 8; ++r)
        S[(r + 8 * hf) * 1024 + t * 128 + wv * 16 + lr] = accs[t][r] * scale;
  }
  __syncthreads();

  // 3. softmax over 1024 cols per row: 16 threads per row, 64 cols each
  const int row = tid >> 4;
  const int sub = tid & 15;
  float* Sr = S + row * 1024 + sub * 64;
  float lmax = -3.0e38f;
  for (int j = 0; j < 64; ++j) lmax = fmaxf(lmax, Sr[j]);
  red[row * 16 + sub] = lmax;
  __syncthreads();
  if (sub == 0) {
    float m = red[row * 16];
    for (int j = 1; j < 16; ++j) m = fmaxf(m, red[row * 16 + j]);
    stat[row] = m;
  }
  __syncthreads();
  const float rmax = stat[row];
  float lsum = 0.0f;
  for (int j = 0; j < 64; ++j) {
    float e = __expf(Sr[j] - rmax);
    Sr[j] = e;
    lsum += e;
  }
  red[row * 16 + sub] = lsum;
  __syncthreads();
  if (sub == 0) {
    float s = 0.0f;
    for (int j = 0; j < 16; ++j) s += red[row * 16 + j];
    stat[row] = 1.0f / s;
  }
  __syncthreads();
  const float rinv = stat[row];
  for (int j = 0; j < 64; ++j) Sr[j] *= rinv;
  __syncthreads();

  // 4. O = P V : wave owns 64 output cols (4 accumulators), K-loop over L=1024
  const int d0 = wv * 64;
  v8f accd[4] = {{}, {}, {}, {}};
  for (int kk = 0; kk < 1024; kk += 32) {
    v16bf af = frag_from_f32(S + (long)lr * 1024 + kk + 8 * hf);
#pragma unroll
    for (int j = 0; j < 4; ++j) {
      const bf16_t* pv = Vt + (b * 512 + d0 + j * 16 + lr) * 1024 + kk + 8 * hf;
      v16bf bf = load_frag_bf16(pv);
      accd[j] = wmma_bf16(af, bf, accd[j]);
    }
  }
#pragma unroll
  for (int j = 0; j < 4; ++j)
#pragma unroll
    for (int r = 0; r < 8; ++r)
      O[(b * 4096 + m0 + r + 8 * hf) * 512 + d0 + j * 16 + lr] =
          (bf16_t)accd[j][r];
}

// ---------------------------------------------------------------------------
extern "C" void kernel_launch(void* const* d_in, const int* in_sizes, int n_in,
                              void* d_out, int out_size, void* d_ws, size_t ws_size,
                              hipStream_t stream) {
  const float* input = (const float*)d_in[0];    // [8,512,64,64]
  const float* context = (const float*)d_in[1];  // [8,512,32,32]
  const float* piw = (const float*)d_in[2];      // [512,512]  (out,in)
  const float* pib = (const float*)d_in[3];
  const float* wq = (const float*)d_in[4];       // [512,512]  (in,out)
  const float* bq = (const float*)d_in[5];
  const float* wk = (const float*)d_in[6];
  const float* bk = (const float*)d_in[7];
  const float* wv = (const float*)d_in[8];
  const float* bv = (const float*)d_in[9];
  const float* pow_ = (const float*)d_in[10];    // [512,1024] (out,in)
  const float* pob = (const float*)d_in[11];
  float* out = (float*)d_out;                    // [8,512,64,64]

  const int B = 8, HW = 4096, L = 1024, E = 512, CIN = 512;

  char* ws = (char*)d_ws;
  size_t off = 0;
  auto alloc = [&](size_t bytes) {
    char* p = ws + off;
    off += (bytes + 255) & ~(size_t)255;
    return p;
  };
  bf16_t* Xin = (bf16_t*)alloc((size_t)B * HW * CIN * 2);  // input, NHWC bf16
  bf16_t* Ctx = (bf16_t*)alloc((size_t)B * L * E * 2);     // context, NLC bf16
  bf16_t* Xp  = (bf16_t*)alloc((size_t)B * HW * E * 2);    // proj_in out; reused as O
  bf16_t* Qb  = (bf16_t*)alloc((size_t)B * HW * E * 2);
  bf16_t* Kb  = (bf16_t*)alloc((size_t)B * L * E * 2);
  bf16_t* Vt  = (bf16_t*)alloc((size_t)B * E * L * 2);     // V transposed [d][l]
  bf16_t* Wpi = (bf16_t*)alloc((size_t)E * CIN * 2);
  bf16_t* Wqt = (bf16_t*)alloc((size_t)E * E * 2);
  bf16_t* Wkt = (bf16_t*)alloc((size_t)E * E * 2);
  bf16_t* Wvt = (bf16_t*)alloc((size_t)E * E * 2);
  bf16_t* W1t = (bf16_t*)alloc((size_t)CIN * CIN * 2);
  bf16_t* W2t = (bf16_t*)alloc((size_t)CIN * E * 2);
  bf16_t* Oatt = Xp;  // Xp dead after Q projection

  // --- pack activations & weights to bf16 ---
  nchw_to_nhwc_bf16<<<dim3((CIN * HW) / 256, B), 256, 0, stream>>>(input, Xin, CIN, HW);
  nchw_to_nhwc_bf16<<<dim3((E * L) / 256, B), 256, 0, stream>>>(context, Ctx, E, L);
  w_convert_bf16<<<dim3((E * CIN) / 256), 256, 0, stream>>>(piw, Wpi, E, CIN, CIN, 0);
  w_transpose_bf16<<<dim3((E * E) / 256), 256, 0, stream>>>(wq, Wqt, E, E);
  w_transpose_bf16<<<dim3((E * E) / 256), 256, 0, stream>>>(wk, Wkt, E, E);
  w_transpose_bf16<<<dim3((E * E) / 256), 256, 0, stream>>>(wv, Wvt, E, E);
  w_convert_bf16<<<dim3((CIN * CIN) / 256), 256, 0, stream>>>(pow_, W1t, CIN, CIN, CIN + E, 0);
  w_convert_bf16<<<dim3((CIN * E) / 256), 256, 0, stream>>>(pow_, W2t, CIN, E, CIN + E, CIN);

  // --- proj_in: Xp = Xin * Wpi^T + pib   [4096x512] per batch ---
  gemm_bf16_kernel<<<dim3(E / 256, HW / 128, B), 256, 0, stream>>>(
      Xin, Wpi, pib, Xp, HW, E, CIN, (long)HW * CIN, 0, (long)HW * E, E,
      FLAG_OUT_BF16);
  // --- Q = Xp * Wq + bq ---
  gemm_bf16_kernel<<<dim3(E / 256, HW / 128, B), 256, 0, stream>>>(
      Xp, Wqt, bq, Qb, HW, E, E, (long)HW * E, 0, (long)HW * E, E,
      FLAG_OUT_BF16);
  // --- K = Ctx * Wk + bk   [1024x512] ---
  gemm_bf16_kernel<<<dim3(E / 256, L / 128, B), 256, 0, stream>>>(
      Ctx, Wkt, bk, Kb, L, E, E, (long)L * E, 0, (long)L * E, E,
      FLAG_OUT_BF16);
  // --- V = Ctx * Wv + bv, stored transposed [512x1024] ---
  gemm_bf16_kernel<<<dim3(E / 256, L / 128, B), 256, 0, stream>>>(
      Ctx, Wvt, bv, Vt, L, E, E, (long)L * E, 0, (long)E * L, L,
      FLAG_OUT_BF16 | FLAG_TRANS);

  // --- attention ---
  const float scale = 1.0f / sqrtf((float)E);
  const size_t smem = 16 * 512 * 2 + 16 * 1024 * 4 + 16 * 16 * 4 + 16 * 4;
  attn_kernel<<<dim3(HW / 16, B), 256, smem, stream>>>(Qb, Kb, Vt, Oatt, scale);

  // --- proj_out: out[b][o][hw] = Xin*W1^T + Oatt*W2^T + pob (transposed f32) ---
  gemm_bf16_kernel<<<dim3(CIN / 256, HW / 128, B), 256, 0, stream>>>(
      Xin, W1t, pob, out, HW, CIN, CIN, (long)HW * CIN, 0, (long)CIN * HW, HW,
      FLAG_TRANS);
  gemm_bf16_kernel<<<dim3(CIN / 256, HW / 128, B), 256, 0, stream>>>(
      Oatt, W2t, nullptr, out, HW, CIN, E, (long)HW * E, 0, (long)CIN * HW, HW,
      FLAG_TRANS | FLAG_ACCUM);

  (void)in_sizes; (void)n_in; (void)out_size; (void)ws_size;
}
